// Qwen3MoeEagerSparseMoeBlock_9225589752213
// MI455X (gfx1250) — matrix operand order repaired and checked by reading, not compiled, not run on hardware
//
#include <hip/hip_runtime.h>
#include <hip/hip_bf16.h>

#define T_TOKENS 2048
#define HID 2048
#define INTER 768
#define NEXP 8

typedef __attribute__((ext_vector_type(16))) __bf16 v16bf;
typedef __attribute__((ext_vector_type(2))) __bf16 v2bf;
typedef __attribute__((ext_vector_type(8))) float v8f;

union Frag16 { v16bf v; unsigned int u[8]; };

// f32 -> bf16 via hardware convert (RNE), packed pair -> one dword
__device__ __forceinline__ unsigned int pack2bf(float lo, float hi) {
#if __has_builtin(__builtin_amdgcn_cvt_pk_bf16_f32)
  auto p = __builtin_amdgcn_cvt_pk_bf16_f32(lo, hi);
  return __builtin_bit_cast(unsigned int, p);
#else
  v2bf p;
  p.x = (__bf16)lo;
  p.y = (__bf16)hi;
  return __builtin_bit_cast(unsigned int, p);
#endif
}

__device__ __forceinline__ unsigned short f2bf(float f) {
  return __builtin_bit_cast(unsigned short, (__bf16)f);
}

__device__ __forceinline__ v8f wmma_bf16(const Frag16& a, const Frag16& b, v8f c) {
  return __builtin_amdgcn_wmma_f32_16x16x32_bf16(false, a.v, false, b.v,
                                                 (short)0, c, false, false);
}

// ---------------------------------------------------------------- utilities
__global__ void zero_init_kernel(float* __restrict__ out, int n,
                                 int* __restrict__ counts) {
  int i = blockIdx.x * blockDim.x + threadIdx.x;
  int stride = gridDim.x * blockDim.x;
  for (; i < n; i += stride) out[i] = 0.0f;
  if (blockIdx.x == 0 && threadIdx.x < NEXP) counts[threadIdx.x] = 0;
}

// n4 = n/4 (T*H is a multiple of 4)
__global__ void cvt_x_kernel(const float* __restrict__ x,
                             unsigned int* __restrict__ xb, int n4) {
  int i = blockIdx.x * blockDim.x + threadIdx.x;
  int stride = gridDim.x * blockDim.x;
  for (; i < n4; i += stride) {
    float4 v = ((const float4*)x)[i];
    uint2 p;
    p.x = pack2bf(v.x, v.y);
    p.y = pack2bf(v.z, v.w);
    ((uint2*)xb)[i] = p;
  }
}

// ---------------------------------------------------------------- router
__global__ __launch_bounds__(256) void router_kernel(
    const float* __restrict__ x, const float* __restrict__ Wgate,
    int* __restrict__ tidx, float* __restrict__ tw) {
  const int t = blockIdx.x;
  const float* xr = x + (size_t)t * HID;
  __shared__ float red[NEXP][256];
  float acc[NEXP];
#pragma unroll
  for (int e = 0; e < NEXP; ++e) acc[e] = 0.0f;
  for (int h = threadIdx.x; h < HID; h += 256) {
    float xv = xr[h];
#pragma unroll
    for (int e = 0; e < NEXP; ++e) acc[e] += xv * Wgate[e * HID + h];
  }
#pragma unroll
  for (int e = 0; e < NEXP; ++e) red[e][threadIdx.x] = acc[e];
  __syncthreads();
  for (int off = 128; off > 0; off >>= 1) {
    if (threadIdx.x < off) {
#pragma unroll
      for (int e = 0; e < NEXP; ++e)
        red[e][threadIdx.x] += red[e][threadIdx.x + off];
    }
    __syncthreads();
  }
  if (threadIdx.x == 0) {
    float l[NEXP], p[NEXP];
    float mx = -3.4e38f;
#pragma unroll
    for (int e = 0; e < NEXP; ++e) { l[e] = red[e][0]; mx = fmaxf(mx, l[e]); }
#pragma unroll
    for (int e = 0; e < NEXP; ++e) p[e] = __expf(l[e] - mx);
    int i1 = 0;
#pragma unroll
    for (int e = 1; e < NEXP; ++e) if (p[e] > p[i1]) i1 = e;  // first max wins
    int i2 = (i1 == 0) ? 1 : 0;
#pragma unroll
    for (int e = 0; e < NEXP; ++e) if (e != i1 && p[e] > p[i2]) i2 = e;
    float s2 = p[i1] + p[i2];
    tidx[t * 2 + 0] = i1;
    tidx[t * 2 + 1] = i2;
    tw[t * 2 + 0] = p[i1] / s2;
    tw[t * 2 + 1] = p[i2] / s2;
  }
}

// ---------------------------------------------------------------- dispatch
__global__ void build_lists_kernel(const int* __restrict__ tidx,
                                   int* __restrict__ counts,
                                   int* __restrict__ lists) {
  int id = blockIdx.x * blockDim.x + threadIdx.x;
  if (id >= T_TOKENS * 2) return;
  int e = tidx[id];
  int pos = atomicAdd(&counts[e], 1);
  lists[e * T_TOKENS + pos] = id;  // encodes token*2 + k
}

// ------------------------------------------------------- gate/up + SiLU*mul
// block tile: 128 tokens x 64 inter cols, 8 waves, K-chunks of 32,
// double-buffered LDS with load/store split for global/WMMA overlap.
__global__ __launch_bounds__(256) void moe_gemm1(
    const unsigned short* __restrict__ xb, const float* __restrict__ Wg,
    const float* __restrict__ Wu, const int* __restrict__ counts,
    const int* __restrict__ lists, unsigned short* __restrict__ hb) {
  const int e = blockIdx.z;
  const int cnt = counts[e];
  const int rowBase = blockIdx.y * 128;
  if (rowBase >= cnt) return;
  const int n0 = blockIdx.x * 64;
  __shared__ unsigned short As[2][128 * 32];
  __shared__ unsigned short Bgs[2][64 * 32];  // [n][k] transposed
  __shared__ unsigned short Bus[2][64 * 32];
  __shared__ int rowEnt[128];
  const int tid = threadIdx.x;
  if (tid < 128) {
    int r = rowBase + tid;
    rowEnt[tid] = (r < cnt) ? lists[e * T_TOKENS + r] : -1;
  }
  __syncthreads();
  const int lane = tid & 31;
  const int wv = tid >> 5;        // 0..7
  const int khalf = lane >> 4;
  const int l15 = lane & 15;
  v8f cg[4] = {};
  v8f cu[4] = {};
  const float* wgE = Wg + (size_t)e * HID * INTER;
  const float* wuE = Wu + (size_t)e * HID * INTER;
  // staging assignment
  const int ar = tid >> 1;          // A row 0..127
  const int ah = (tid & 1) * 16;    // A k-offset (16 bf16 = 32B)
  const int bkp = tid >> 4;         // B k-pair 0..15 (k = 2*bkp)
  const int bn4 = (tid & 15) * 4;   // B n base (4 cols)
  const int aent = rowEnt[ar];

  uint4 a0, a1;
  float4 g0, g1, u0, u1;

  auto load = [&](int k0) {
    a0 = make_uint4(0, 0, 0, 0);
    a1 = a0;
    if (aent >= 0) {
      const uint4* s = (const uint4*)(xb + (size_t)(aent >> 1) * HID + k0 + ah);
      a0 = s[0];
      a1 = s[1];
    }
    const float* pg = wgE + (size_t)(k0 + bkp * 2) * INTER + n0 + bn4;
    const float* pu = wuE + (size_t)(k0 + bkp * 2) * INTER + n0 + bn4;
    g0 = *(const float4*)pg;
    g1 = *(const float4*)(pg + INTER);
    u0 = *(const float4*)pu;
    u1 = *(const float4*)(pu + INTER);
  };
  auto store = [&](int buf) {
    uint4* d = (uint4*)&As[buf][ar * 32 + ah];
    d[0] = a0;
    d[1] = a1;
    float ga[4] = {g0.x, g0.y, g0.z, g0.w};
    float gb[4] = {g1.x, g1.y, g1.z, g1.w};
    float ua[4] = {u0.x, u0.y, u0.z, u0.w};
    float ub[4] = {u1.x, u1.y, u1.z, u1.w};
#pragma unroll
    for (int i = 0; i < 4; ++i) {
      ((unsigned int*)&Bgs[buf][(bn4 + i) * 32])[bkp] = pack2bf(ga[i], gb[i]);
      ((unsigned int*)&Bus[buf][(bn4 + i) * 32])[bkp] = pack2bf(ua[i], ub[i]);
    }
  };
  auto compute = [&](int buf) {
    Frag16 a;
    const unsigned int* arow =
        (const unsigned int*)&As[buf][(wv * 16 + l15) * 32];
#pragma unroll
    for (int j = 0; j < 4; ++j) {  // ISA 16-bit A 16x32 interleave
      a.u[j] = arow[khalf * 4 + j];
      a.u[4 + j] = arow[8 + khalf * 4 + j];
    }
#pragma unroll
    for (int s = 0; s < 4; ++s) {
      Frag16 bg, bu;
      const unsigned int* bgrow =
          (const unsigned int*)&Bgs[buf][(s * 16 + l15) * 32];
      const unsigned int* burow =
          (const unsigned int*)&Bus[buf][(s * 16 + l15) * 32];
#pragma unroll
      for (int j = 0; j < 8; ++j) {
        bg.u[j] = bgrow[(j >> 2) * 8 + khalf * 4 + (j & 3)];
        bu.u[j] = burow[(j >> 2) * 8 + khalf * 4 + (j & 3)];
      }
      cg[s] = wmma_bf16(a, bg, cg[s]);
      cu[s] = wmma_bf16(a, bu, cu[s]);
    }
  };

  load(0);
  for (int k0 = 0; k0 < HID; k0 += 32) {
    const int buf = (k0 >> 5) & 1;
    store(buf);
    __syncthreads();
    if (k0 + 32 < HID) load(k0 + 32);
    compute(buf);
  }
  // epilogue: h = silu(g) * u -> bf16
#pragma unroll
  for (int s = 0; s < 4; ++s) {
#pragma unroll
    for (int v = 0; v < 8; ++v) {
      int m = wv * 16 + khalf * 8 + v;
      int ent = rowEnt[m];
      if (ent >= 0) {
        float g = cg[s][v], uu = cu[s][v];
        float sil = g / (1.0f + __expf(-g));
        hb[(size_t)ent * INTER + n0 + s * 16 + l15] = f2bf(sil * uu);
      }
    }
  }
}

// ------------------------------------------------------- down proj + combine
__global__ __launch_bounds__(256) void moe_gemm2(
    const unsigned short* __restrict__ hb, const float* __restrict__ Wd,
    const int* __restrict__ counts, const int* __restrict__ lists,
    const float* __restrict__ tw, float* __restrict__ out) {
  const int e = blockIdx.z;
  const int cnt = counts[e];
  const int rowBase = blockIdx.y * 128;
  if (rowBase >= cnt) return;
  const int n0 = blockIdx.x * 64;
  __shared__ unsigned short As[2][128 * 32];
  __shared__ unsigned short Bs[2][64 * 32];
  __shared__ int rowEnt[128];
  __shared__ float rowW[128];
  const int tid = threadIdx.x;
  if (tid < 128) {
    int r = rowBase + tid;
    int ent = (r < cnt) ? lists[e * T_TOKENS + r] : -1;
    rowEnt[tid] = ent;
    rowW[tid] = (ent >= 0) ? tw[ent] : 0.0f;
  }
  __syncthreads();
  const int lane = tid & 31;
  const int wv = tid >> 5;
  const int khalf = lane >> 4;
  const int l15 = lane & 15;
  v8f c[4] = {};
  const float* wdE = Wd + (size_t)e * INTER * HID;
  const int ar = tid >> 1;
  const int ah = (tid & 1) * 16;
  const int bkp = tid >> 4;
  const int bn4 = (tid & 15) * 4;
  const int aent = rowEnt[ar];

  uint4 a0, a1;
  float4 b0, b1;

  auto load = [&](int k0) {
    a0 = make_uint4(0, 0, 0, 0);
    a1 = a0;
    if (aent >= 0) {
      const uint4* s = (const uint4*)(hb + (size_t)aent * INTER + k0 + ah);
      a0 = s[0];
      a1 = s[1];
    }
    const float* pb = wdE + (size_t)(k0 + bkp * 2) * HID + n0 + bn4;
    b0 = *(const float4*)pb;
    b1 = *(const float4*)(pb + HID);
  };
  auto store = [&](int buf) {
    uint4* d = (uint4*)&As[buf][ar * 32 + ah];
    d[0] = a0;
    d[1] = a1;
    float ba[4] = {b0.x, b0.y, b0.z, b0.w};
    float bb[4] = {b1.x, b1.y, b1.z, b1.w};
#pragma unroll
    for (int i = 0; i < 4; ++i)
      ((unsigned int*)&Bs[buf][(bn4 + i) * 32])[bkp] = pack2bf(ba[i], bb[i]);
  };
  auto compute = [&](int buf) {
    Frag16 a;
    const unsigned int* arow =
        (const unsigned int*)&As[buf][(wv * 16 + l15) * 32];
#pragma unroll
    for (int j = 0; j < 4; ++j) {
      a.u[j] = arow[khalf * 4 + j];
      a.u[4 + j] = arow[8 + khalf * 4 + j];
    }
#pragma unroll
    for (int s = 0; s < 4; ++s) {
      Frag16 b;
      const unsigned int* brow =
          (const unsigned int*)&Bs[buf][(s * 16 + l15) * 32];
#pragma unroll
      for (int j = 0; j < 8; ++j) b.u[j] = brow[(j >> 2) * 8 + khalf * 4 + (j & 3)];
      c[s] = wmma_bf16(a, b, c[s]);
    }
  };

  load(0);
  for (int k0 = 0; k0 < INTER; k0 += 32) {
    const int buf = (k0 >> 5) & 1;
    store(buf);
    __syncthreads();
    if (k0 + 32 < INTER) load(k0 + 32);
    compute(buf);
  }
  // epilogue: out[t] += w * y (exactly 2 commutative f32 adds -> deterministic)
#pragma unroll
  for (int s = 0; s < 4; ++s) {
#pragma unroll
    for (int v = 0; v < 8; ++v) {
      int m = wv * 16 + khalf * 8 + v;
      int ent = rowEnt[m];
      if (ent >= 0) {
        float y = c[s][v] * rowW[m];
        int t = ent >> 1;
        unsafeAtomicAdd(out + (size_t)t * HID + n0 + s * 16 + l15, y);
      }
    }
  }
}

// ---------------------------------------------------------------- launcher
extern "C" void kernel_launch(void* const* d_in, const int* in_sizes, int n_in,
                              void* d_out, int out_size, void* d_ws,
                              size_t ws_size, hipStream_t stream) {
  (void)in_sizes; (void)n_in; (void)out_size; (void)ws_size;
  const float* x = (const float*)d_in[0];      // [4,512,2048]
  const float* Wgate = (const float*)d_in[1];  // [8,2048]
  const float* Wg = (const float*)d_in[2];     // [8,2048,768]
  const float* Wu = (const float*)d_in[3];     // [8,2048,768]
  const float* Wd = (const float*)d_in[4];     // [8,768,2048]
  float* out = (float*)d_out;

  char* w = (char*)d_ws;
  size_t off = 0;
  auto alloc = [&](size_t bytes) -> void* {
    void* p = w + off;
    off = (off + bytes + 255) & ~(size_t)255;
    return p;
  };
  unsigned short* xb = (unsigned short*)alloc((size_t)T_TOKENS * HID * 2);
  unsigned short* hb = (unsigned short*)alloc((size_t)T_TOKENS * 2 * INTER * 2);
  int* tidx = (int*)alloc((size_t)T_TOKENS * 2 * 4);
  float* tw = (float*)alloc((size_t)T_TOKENS * 2 * 4);
  int* counts = (int*)alloc(NEXP * 4);
  int* lists = (int*)alloc((size_t)NEXP * T_TOKENS * 4);

  const int nOut = T_TOKENS * HID;
  zero_init_kernel<<<1024, 256, 0, stream>>>(out, nOut, counts);
  cvt_x_kernel<<<512, 256, 0, stream>>>(x, (unsigned int*)xb, nOut / 4);
  router_kernel<<<T_TOKENS, 256, 0, stream>>>(x, Wgate, tidx, tw);
  build_lists_kernel<<<(T_TOKENS * 2 + 255) / 256, 256, 0, stream>>>(tidx, counts, lists);
  moe_gemm1<<<dim3(INTER / 64, T_TOKENS / 128, NEXP), 256, 0, stream>>>(
      xb, Wg, Wu, counts, lists, hb);
  moe_gemm2<<<dim3(HID / 64, T_TOKENS / 128, NEXP), 256, 0, stream>>>(
      hb, Wd, counts, lists, tw, out);
}
